// Edge_Feat_Init_15607911153855
// MI455X (gfx1250) — matrix operand order repaired and checked by reading, not compiled, not run on hardware
//
#include <hip/hip_runtime.h>
#include <hip/hip_bf16.h>

typedef __attribute__((ext_vector_type(16))) _Float16     v16h;
typedef __attribute__((ext_vector_type(8)))  float        v8f;
typedef __attribute__((ext_vector_type(4)))  unsigned int v4u;

#define NUM_RBF 50
#define HIDDEN  128
#define KPAD    64     // K padded 50 -> 64 (two WMMA k-steps of 32)
#define CUT_HI  5.0f
#define PI_F    3.14159265358979323846f

union Frag { v16h h; v4u u[2]; };

// ---------------------------------------------------------------------------
// Main kernel: full 16-edge tiles only. 4 waves/block, one tile per wave.
// ---------------------------------------------------------------------------
__global__ __launch_bounds__(128)
void edge_feat_wmma_kernel(const float* __restrict__ pos,
                           const long long* __restrict__ ei,
                           const float* __restrict__ means,
                           const float* __restrict__ betas,
                           const float* __restrict__ W,
                           const float* __restrict__ bias,
                           float* __restrict__ out,
                           long long E,
                           long long ntiles)
{
    __shared__ __attribute__((aligned(16))) _Float16 sW[HIDDEN * KPAD];   // 16 KB
    __shared__ __attribute__((aligned(16))) _Float16 sT[4][16 * KPAD];    //  8 KB
    __shared__ float sBias[HIDDEN];
    __shared__ float sMean[KPAD];
    __shared__ float sBeta[KPAD];

    const int tid = threadIdx.x;

    // ---- Phase 0: stage W (f16, padded), bias, means, betas into LDS ----
    {
        const float* wrow = W + tid * NUM_RBF;      // 128 threads -> 128 rows
        _Float16*    drow = sW + tid * KPAD;
        #pragma unroll
        for (int k = 0; k < KPAD; ++k)
            drow[k] = (k < NUM_RBF) ? (_Float16)wrow[k] : (_Float16)0.0f;
        sBias[tid] = bias[tid];
        if (tid < KPAD) {
            sMean[tid] = (tid < NUM_RBF) ? means[tid] : 0.0f;
            sBeta[tid] = (tid < NUM_RBF) ? betas[tid] : 0.0f;
        }
    }
    __syncthreads();   // only block-wide barrier needed (sW/sBias staging)

    const int wave = tid >> 5;
    const int lane = tid & 31;
    const int m    = lane & 15;   // edge-in-tile (matrix row M)
    const int hi   = lane >> 4;   // lane half selector

    const long long tile = (long long)blockIdx.x * 4 + wave;
    if (tile >= ntiles) return;   // idle wave in partial last block

    const long long e0 = tile * 16;
    const long long e  = e0 + m;  // always < E (full tiles only)

    // Prefetch next tile's indices into cache hierarchy (global_prefetch_b8).
    __builtin_prefetch(ei + e + 64, 0, 0);

    const long long s = ei[e];
    const long long t = ei[E + e];
    const float dx = pos[3*s+0] - pos[3*t+0];
    const float dy = pos[3*s+1] - pos[3*t+1];
    const float dz = pos[3*s+2] - pos[3*t+2];
    const float d2   = dx*dx + dy*dy + dz*dz;
    const float d    = sqrtf(d2);
    const float invd = __frsqrt_rn(d2);
    const float cosv = __cosf(d * (PI_F / CUT_HI));
    const float cutoff = (d < CUT_HI) ? 0.5f * (cosv + 1.0f) : 0.0f;
    const float ex   = __expf(-d);   // exp(alpha*(CUT_LO - d)), alpha = 1, CUT_LO = 0

    // Output layout (flat, return order): idx(2E) | weight(E) | attr(128E) | vec(3E)
    float* out_idx  = out;
    float* out_w    = out + 2*E;
    float* out_attr = out + 3*E;
    float* out_vec  = out + 131*E;

    if (hi == 0) {
        __builtin_nontemporal_store((float)s,  out_idx + e);
        __builtin_nontemporal_store((float)t,  out_idx + E + e);
        __builtin_nontemporal_store(d,         out_w + e);
        __builtin_nontemporal_store(dx * invd, out_vec + 3*e + 0);
        __builtin_nontemporal_store(dy * invd, out_vec + 3*e + 1);
        __builtin_nontemporal_store(dz * invd, out_vec + 3*e + 2);
    }

    // ---- Phase 1: RBF -> f16 LDS tile (row m, K = hi*32 .. hi*32+31) ----
    // sT[wave] is wave-private; wave LDS ops are in-order, no barrier needed.
    {
        _Float16* trow = sT[wave] + m * KPAD;
        const int kb = hi * 32;
        #pragma unroll
        for (int kk = 0; kk < 32; kk += 2) {
            const int k0 = kb + kk, k1 = k0 + 1;
            const float t0 = ex - sMean[k0];
            const float t1 = ex - sMean[k1];
            const float r0 = (k0 < NUM_RBF) ? cutoff * __expf(-sBeta[k0]*t0*t0) : 0.0f;
            const float r1 = (k1 < NUM_RBF) ? cutoff * __expf(-sBeta[k1]*t1*t1) : 0.0f;
            union { _Float16 h[2]; unsigned u; } p;
            p.h[0] = (_Float16)r0; p.h[1] = (_Float16)r1;
            *(unsigned*)(trow + k0) = p.u;             // ds_store_b32
        }
    }

    // ---- Phase 2: 16x128 = rbf(16x64) @ Wt(64x128) via 16 WMMAs ----
    // A 16-bit 16x32 layout: lanes 0-15 hold K 0-7 & 16-23, lanes 16-31 hold
    // K 8-15 & 24-31 of row M=lane&15 -> two ds_load_b128 per k-step.
    Frag a0, a1;
    {
        const _Float16* trow = sT[wave] + m * KPAD;
        a0.u[0] = *(const v4u*)(trow +  0 + hi*8);
        a0.u[1] = *(const v4u*)(trow + 16 + hi*8);
        a1.u[0] = *(const v4u*)(trow + 32 + hi*8);
        a1.u[1] = *(const v4u*)(trow + 48 + hi*8);
    }

    // Base pointer for this lane's C rows: M = r + 8*hi, N = m (+ nt*16).
    float* pbase = out_attr + (size_t)(e0 + 8*hi) * HIDDEN + m;

    #pragma unroll
    for (int nt = 0; nt < 8; ++nt) {
        const int n = nt * 16 + m;                      // output feature column
        const _Float16* wrow = sW + n * KPAD;
        // B 32x16: lane holds column N=lane&15, K = kbase + hi*8 .. +7 and kbase + 16 + hi*8 .. +7
        Frag b0, b1;
        b0.u[0] = *(const v4u*)(wrow +  0 + hi*8);
        b0.u[1] = *(const v4u*)(wrow + 16 + hi*8);
        b1.u[0] = *(const v4u*)(wrow + 32 + hi*8);
        b1.u[1] = *(const v4u*)(wrow + 48 + hi*8);

        v8f c = {};
        c = __builtin_amdgcn_wmma_f32_16x16x32_f16(false, a0.h, false, b0.h,
                                                   (short)0, c, false, false);
        c = __builtin_amdgcn_wmma_f32_16x16x32_f16(false, a1.h, false, b1.h,
                                                   (short)0, c, false, false);

        const float bn = sBias[n];
        float* p = pbase + nt * 16;
        // Unconditional NT stores at constant immediate offsets (r*512 B).
        #pragma unroll
        for (int r = 0; r < 8; ++r)
            __builtin_nontemporal_store(c[r] + bn, p + r * HIDDEN);
    }
}

// ---------------------------------------------------------------------------
// Tail kernel: one block per leftover edge (E % 16 edges; never for E=1.6M).
// ---------------------------------------------------------------------------
__global__ __launch_bounds__(128)
void edge_feat_tail_kernel(const float* __restrict__ pos,
                           const long long* __restrict__ ei,
                           const float* __restrict__ means,
                           const float* __restrict__ betas,
                           const float* __restrict__ W,
                           const float* __restrict__ bias,
                           float* __restrict__ out,
                           long long E,
                           long long e_base)
{
    __shared__ float srbf[NUM_RBF];

    const long long e = e_base + blockIdx.x;
    const int tid = threadIdx.x;

    const long long s = ei[e];
    const long long t = ei[E + e];
    const float dx = pos[3*s+0] - pos[3*t+0];
    const float dy = pos[3*s+1] - pos[3*t+1];
    const float dz = pos[3*s+2] - pos[3*t+2];
    const float d2   = dx*dx + dy*dy + dz*dz;
    const float d    = sqrtf(d2);
    const float invd = __frsqrt_rn(d2);
    const float cosv = __cosf(d * (PI_F / CUT_HI));
    const float cutoff = (d < CUT_HI) ? 0.5f * (cosv + 1.0f) : 0.0f;
    const float ex   = __expf(-d);

    if (tid < NUM_RBF) {
        const float tk = ex - means[tid];
        srbf[tid] = cutoff * __expf(-betas[tid] * tk * tk);
    }
    __syncthreads();

    float* out_idx  = out;
    float* out_w    = out + 2*E;
    float* out_attr = out + 3*E;
    float* out_vec  = out + 131*E;

    if (tid == 0) {
        out_idx[e]       = (float)s;
        out_idx[E + e]   = (float)t;
        out_w[e]         = d;
        out_vec[3*e + 0] = dx * invd;
        out_vec[3*e + 1] = dy * invd;
        out_vec[3*e + 2] = dz * invd;
    }

    // One output feature per thread.
    const float* wrow = W + tid * NUM_RBF;
    float acc = bias[tid];
    #pragma unroll
    for (int k = 0; k < NUM_RBF; ++k)
        acc += srbf[k] * wrow[k];
    out_attr[e * HIDDEN + tid] = acc;
}

extern "C" void kernel_launch(void* const* d_in, const int* in_sizes, int n_in,
                              void* d_out, int out_size, void* d_ws, size_t ws_size,
                              hipStream_t stream) {
    const float*     pos   = (const float*)d_in[0];
    const long long* ei    = (const long long*)d_in[1];
    const float*     means = (const float*)d_in[2];
    const float*     betas = (const float*)d_in[3];
    const float*     W     = (const float*)d_in[4];
    const float*     bias  = (const float*)d_in[5];
    float*           out   = (float*)d_out;

    const long long E      = (long long)in_sizes[1] / 2;   // edge_index is (2, E)
    const long long ntiles = E / 16;                       // full 16-edge tiles
    const long long rem    = E - ntiles * 16;              // ragged tail (< 16)

    if (ntiles > 0) {
        const int grid = (int)((ntiles + 3) / 4);          // 4 tiles (waves) per block
        edge_feat_wmma_kernel<<<grid, 128, 0, stream>>>(
            pos, ei, means, betas, W, bias, out, E, ntiles);
    }
    if (rem > 0) {
        edge_feat_tail_kernel<<<(int)rem, 128, 0, stream>>>(
            pos, ei, means, betas, W, bias, out, E, ntiles * 16);
    }
}
